// SwinBlock_36163624632922
// MI455X (gfx1250) — matrix operand order repaired and checked by reading, not compiled, not run on hardware
//
#include <hip/hip_runtime.h>
#include <hip/hip_bf16.h>
#include <math.h>

typedef __attribute__((ext_vector_type(16))) __bf16 v16bf;
typedef __attribute__((ext_vector_type(8)))  float  v8f;

#define WMMA_BF16(a, b, c) \
  __builtin_amdgcn_wmma_f32_16x16x32_bf16(false, (a), false, (b), (short)0, (c), false, false)

__device__ __forceinline__ v8f v8f_zero() {
  v8f z = {0.f, 0.f, 0.f, 0.f, 0.f, 0.f, 0.f, 0.f};
  return z;
}

// ---------------------------------------------------------------------------
// Weight pre-swizzle: f32 [Cin, Cout] row-major -> bf16 WMMA B-operand panels:
// layout [kt][nt][lane(32)][e(16)], element = W[kt*32 + (e>=8)*16 + (lane>=16)*8 + e%8, nt*16 + lane%16]
// ---------------------------------------------------------------------------
__global__ void swizzle_b_kernel(const float* __restrict__ W, __bf16* __restrict__ dst,
                                 int Cout, int ntn, int total) {
  int i = blockIdx.x * 256 + threadIdx.x;
  if (i >= total) return;
  int e    = i & 15;
  int lane = (i >> 4) & 31;
  int nt   = (i >> 9) % ntn;
  int kt   = i / (512 * ntn);
  int k = kt * 32 + ((e >> 3) << 4) + ((lane >> 4) << 3) + (e & 7);
  int n = nt * 16 + (lane & 15);
  dst[i] = (__bf16)W[k * Cout + n];
}

// ---------------------------------------------------------------------------
// Kernel A: shifted-window gather + LN1 + QKV + attention + proj -> d_out (xr)
// One workgroup (8 wave32) per window. All activations live in LDS.
// ---------------------------------------------------------------------------
__global__ __launch_bounds__(256) void swin_attn_kernel(
    const float* __restrict__ x,
    const __bf16* __restrict__ Wqkv, const float* __restrict__ bqkv,
    const __bf16* __restrict__ Wproj, const float* __restrict__ bproj,
    const float* __restrict__ n1g, const float* __restrict__ n1b,
    const float* __restrict__ amask,
    float* __restrict__ out) {
  extern __shared__ __align__(32) char smem[];
  float*  lnbuf = (float*)smem;                 // 64*192*4 = 49152 B (reused as Sbuf 64x64 f32)
  __bf16* xnA = (__bf16*)(smem + 49152);        // 24576 B : LN1 out, A-swizzled [4mt][6kt][32][16]
  __bf16* qA  = (__bf16*)(smem + 73728);        // 24576 B : q  A-swizzled [6h][4mt][32][16]
  __bf16* kB  = (__bf16*)(smem + 98304);        // 24576 B : k  B-swizzled [6h][4nt_m][32][16]
  __bf16* vB  = (__bf16*)(smem + 122880);       // 24576 B : v  B-swizzled [6h][2ktm][2dt][32][16]
  __bf16* pA  = (__bf16*)(smem + 147456);       //  8192 B : probs A-swizzled [4mt][2kt][32][16]
  __bf16* oA  = (__bf16*)(smem + 155648);       // 24576 B : attn out A-swizzled [4mt][6kt][32][16]
  // total 180224 B LDS

  const int tid = threadIdx.x;
  const int wv  = tid >> 5;
  const int L   = tid & 31;
  const int win = blockIdx.x;
  const int b   = win >> 6;
  const int w   = win & 63;
  const int wy  = w >> 3, wx = w & 7;

  // Warm L2/WGP$ with the weight panels used by this kernel (qkv + proj).
  for (int i = tid; i < 2304; i += 256)
    __builtin_prefetch((const char*)Wqkv + i * 128, 0, 1);

  // ---- Phase 0: shifted-window gather + LayerNorm1 (f32, wave per token) ----
  for (int t = wv; t < 49; t += 8) {
    int ty = t / 7, tx = t % 7;
    int h2 = (wy * 7 + ty + 3) % 56;
    int w2 = (wx * 7 + tx + 3) % 56;
    const float* src = x + ((b * 3136) + h2 * 56 + w2) * 192;
    float v[6], s = 0.f, s2 = 0.f;
#pragma unroll
    for (int j = 0; j < 6; ++j) { v[j] = src[L + 32 * j]; s += v[j]; s2 += v[j] * v[j]; }
#pragma unroll
    for (int o = 16; o > 0; o >>= 1) { s += __shfl_xor(s, o, 32); s2 += __shfl_xor(s2, o, 32); }
    float mu = s * (1.f / 192.f);
    float rs = rsqrtf(s2 * (1.f / 192.f) - mu * mu + 1e-5f);
#pragma unroll
    for (int j = 0; j < 6; ++j) {
      int c = L + 32 * j;
      lnbuf[t * 192 + c] = (v[j] - mu) * rs * n1g[c] + n1b[c];
    }
  }
  for (int i = tid; i < 15 * 192; i += 256) lnbuf[49 * 192 + i] = 0.f;  // zero pad rows 49..63
  __syncthreads();

  // repack LN1 result -> bf16 A-operand swizzle
  for (int ch = tid; ch < 768; ch += 256) {     // ch = (mt*6 + kt)*32 + lane
    int ln = ch & 31;
    int kt = (ch >> 5) % 6;
    int mt = ch / 192;
    int m = mt * 16 + (ln & 15);
    int base = kt * 32 + ((ln >> 4) << 3);
    const float* srow = lnbuf + m * 192;
    __bf16* dst = xnA + ch * 16;
#pragma unroll
    for (int e = 0; e < 8; ++e) dst[e] = (__bf16)srow[base + e];
#pragma unroll
    for (int e = 0; e < 8; ++e) dst[8 + e] = (__bf16)srow[base + 16 + e];
  }
  __syncthreads();

  // ---- Phase 1: QKV GEMM [64x192] @ [192x576], scatter q/k/v to WMMA layouts.
  // Output tile columns: nt 0..11 -> q, 12..23 -> k, 24..35 -> v (uniform per wave).
  for (int idx = wv * 18; idx < wv * 18 + 18; ++idx) {   // 144 output tiles
    int mt = idx / 36, nt = idx % 36;
    v8f acc = v8f_zero();
#pragma unroll
    for (int kt = 0; kt < 6; ++kt) {
      v16bf a  = *(const v16bf*)(xnA + ((mt * 6 + kt) * 32 + L) * 16);
      v16bf bb = *(const v16bf*)(Wqkv + ((kt * 36 + nt) * 32 + L) * 16);
      acc = WMMA_BF16(a, bb, acc);
    }
    int c = nt * 16 + (L & 15);
    float bias = bqkv[c];
    int hi8 = (L >> 4) << 3;
    if (nt < 12) {                    // q: A-layout, pre-scaled by 1/sqrt(HD)
      int h = c >> 5, d = c & 31, rr = d & 15;
      int e = (rr & 7) + ((d >= 16) ? 8 : 0);
      int laneAdd = (rr >= 8) ? 16 : 0;
      __bf16* dst = qA + ((h * 4 + mt) * 32 + laneAdd) * 16 + e;
#pragma unroll
      for (int r = 0; r < 8; ++r)
        dst[(r + hi8) * 16] = (__bf16)((acc[r] + bias) * 0.17677669529663687f);
    } else if (nt < 24) {             // k: B-layout (K = head dim)
      int cc = c - 192, h = cc >> 5, d = cc & 31;
      int laneAdd = (d & 8) ? 16 : 0;
      __bf16* dst = kB + ((h * 4 + mt) * 32 + laneAdd) * 16 + (d & 7) + ((d >= 16) ? 8 : 0);
#pragma unroll
      for (int r = 0; r < 8; ++r)
        dst[(r + hi8) * 16] = (__bf16)(acc[r] + bias);
    } else {                          // v: B-layout (K = token index)
      int cc = c - 384, h = cc >> 5, d = cc & 31;
      int laneAdd = hi8 << 1;                       // (m&15)>=8 <=> hi8 != 0
      int ktm = mt >> 1;                            // m>>5 is tile-uniform
      __bf16* dst = vB + (((h * 2 + ktm) * 2 + (d >> 4)) * 32 + (d & 15) + laneAdd) * 16;
#pragma unroll
      for (int r = 0; r < 8; ++r)
        dst[r + ((mt & 1) << 3)] = (__bf16)(acc[r] + bias);     // e = (m&7) + 8*((m&31)>=16)
    }
  }
  __syncthreads();

  // ---- Phase 2: per-head attention ----
  float* Sbuf = lnbuf;               // reuse: 64x64 f32 scores
  const float* wm = amask + w * 2401;
  for (int h = 0; h < 6; ++h) {
    // S = (q*scale) @ k^T : 16 tiles, 2 per wave
#pragma unroll
    for (int u = 0; u < 2; ++u) {
      int tix = wv * 2 + u;
      int i = tix >> 2, j = tix & 3;
      v16bf a  = *(const v16bf*)(qA + ((h * 4 + i) * 32 + L) * 16);
      v16bf bb = *(const v16bf*)(kB + ((h * 4 + j) * 32 + L) * 16);
      v8f acc = v8f_zero();
      acc = WMMA_BF16(a, bb, acc);
      int col = j * 16 + (L & 15);
      int hi8 = (L >> 4) << 3;
#pragma unroll
      for (int r = 0; r < 8; ++r) Sbuf[(i * 16 + r + hi8) * 64 + col] = acc[r];
    }
    __syncthreads();
    // masked softmax over rows (wave handles 8 rows; lane covers cols L and L+32)
    for (int n = wv * 8; n < wv * 8 + 8; ++n) {
      int m0 = L, m1 = L + 32;
      float s0 = Sbuf[n * 64 + m0], s1 = Sbuf[n * 64 + m1];
      if (n < 49) {
        if (m0 < 49) s0 += wm[n * 49 + m0];
        if (m1 < 49) s1 += wm[n * 49 + m1];
      }
      if (m0 >= 49) s0 = -1e30f;
      if (m1 >= 49) s1 = -1e30f;
      float mx = fmaxf(s0, s1);
#pragma unroll
      for (int o = 16; o > 0; o >>= 1) mx = fmaxf(mx, __shfl_xor(mx, o, 32));
      float e0 = __expf(s0 - mx), e1 = __expf(s1 - mx);
      float sm = e0 + e1;
#pragma unroll
      for (int o = 16; o > 0; o >>= 1) sm += __shfl_xor(sm, o, 32);
      float inv = 1.f / sm;
      Sbuf[n * 64 + m0] = e0 * inv;
      Sbuf[n * 64 + m1] = e1 * inv;
    }
    __syncthreads();
    // repack probs -> pA (256 chunks, one per thread)
    {
      int ln = tid & 31, kt = (tid >> 5) & 1, mt = tid >> 6;
      int m = mt * 16 + (ln & 15);
      int base = kt * 32 + ((ln >> 4) << 3);
      const float* srow = Sbuf + m * 64;
      __bf16* dst = pA + tid * 16;
#pragma unroll
      for (int e = 0; e < 8; ++e) dst[e] = (__bf16)srow[base + e];
#pragma unroll
      for (int e = 0; e < 8; ++e) dst[8 + e] = (__bf16)srow[base + 16 + e];
    }
    __syncthreads();
    // O = P @ V : 8 tiles, one per wave; scatter straight into proj A-operand
    {
      int mt = wv >> 1, dt = wv & 1;
      v8f acc = v8f_zero();
#pragma unroll
      for (int kt = 0; kt < 2; ++kt) {
        v16bf a  = *(const v16bf*)(pA + ((mt * 2 + kt) * 32 + L) * 16);
        v16bf bb = *(const v16bf*)(vB + (((h * 2 + kt) * 2 + dt) * 32 + L) * 16);
        acc = WMMA_BF16(a, bb, acc);
      }
      int d = dt * 16 + (L & 15);
      int rr = d & 15;
      int e = (rr & 7) + ((d >= 16) ? 8 : 0);
      int laneAdd = (rr >= 8) ? 16 : 0;
      int hi8 = (L >> 4) << 3;
      __bf16* dst = oA + ((mt * 6 + h) * 32 + laneAdd) * 16 + e;
#pragma unroll
      for (int r = 0; r < 8; ++r)
        dst[(r + hi8) * 16] = (__bf16)acc[r];       // m&15 = r + hi8
    }
    __syncthreads();
  }

  // ---- Phase 3: proj GEMM [64x192]@[192x192] + bias, reverse-shift scatter ----
  for (int idx = wv * 6; idx < wv * 6 + 6; ++idx) {      // 48 tiles
    int mt = idx / 12, nt = idx % 12;
    v8f acc = v8f_zero();
#pragma unroll
    for (int kt = 0; kt < 6; ++kt) {
      v16bf a  = *(const v16bf*)(oA + ((mt * 6 + kt) * 32 + L) * 16);
      v16bf bb = *(const v16bf*)(Wproj + ((kt * 12 + nt) * 32 + L) * 16);
      acc = WMMA_BF16(a, bb, acc);
    }
    int c = nt * 16 + (L & 15);
    float bias = bproj[c];
    int hi8 = (L >> 4) << 3;
#pragma unroll
    for (int r = 0; r < 8; ++r) {
      int m = mt * 16 + r + hi8;
      if (m < 49) {
        int ty = m / 7, tx = m % 7;
        int h2 = (wy * 7 + ty + 3) % 56;
        int w2 = (wx * 7 + tx + 3) % 56;
        out[((b * 3136) + h2 * 56 + w2) * 192 + c] = acc[r] + bias;
      }
    }
  }
}

// ---------------------------------------------------------------------------
// Kernel B: LN2 + MLP (fc1+GELU, fc2) + residual, in-place on d_out.
// One workgroup per 64-token chunk (100352/64 = 1568 blocks).
// ---------------------------------------------------------------------------
__global__ __launch_bounds__(256) void swin_mlp_kernel(
    float* __restrict__ io,
    const __bf16* __restrict__ W1, const float* __restrict__ b1,
    const __bf16* __restrict__ W2, const float* __restrict__ b2,
    const float* __restrict__ n2g, const float* __restrict__ n2b) {
  extern __shared__ __align__(32) char smem[];
  float*  xrbuf = (float*)smem;                 // 49152 B : residual copy
  __bf16* hA = (__bf16*)(smem + 49152);         // 24576 B : LN2 out [4mt][6kt][32][16]
  __bf16* gA = (__bf16*)(smem + 73728);         // 98304 B : gelu(fc1) [4mt][24kt][32][16]
  // total 172032 B LDS

  const int tid = threadIdx.x, wv = tid >> 5, L = tid & 31;
  const int base_tok = blockIdx.x * 64;

  // Warm L2/WGP$ with fc1+fc2 panels (589824 B = 4608 cachelines).
  for (int i = tid; i < 4608; i += 256)
    __builtin_prefetch((const char*)W1 + i * 128, 0, 1);

  // ---- LN2 (wave per row), write residual copy + bf16 A-panels ----
  for (int m = wv; m < 64; m += 8) {
    const float* src = io + (base_tok + m) * 192;
    float v[6], s = 0.f, s2 = 0.f;
#pragma unroll
    for (int j = 0; j < 6; ++j) {
      v[j] = src[L + 32 * j];
      s += v[j]; s2 += v[j] * v[j];
      xrbuf[m * 192 + L + 32 * j] = v[j];
    }
#pragma unroll
    for (int o = 16; o > 0; o >>= 1) { s += __shfl_xor(s, o, 32); s2 += __shfl_xor(s2, o, 32); }
    float mu = s * (1.f / 192.f);
    float rs = rsqrtf(s2 * (1.f / 192.f) - mu * mu + 1e-5f);
    int mt = m >> 4;
#pragma unroll
    for (int j = 0; j < 6; ++j) {
      int c = L + 32 * j;
      float hv = (v[j] - mu) * rs * n2g[c] + n2b[c];
      int kt = c >> 5, kl = c & 31, rr = kl & 15;
      int lane2 = (m & 15) + ((rr >= 8) ? 16 : 0);
      int e = (rr & 7) + ((kl >= 16) ? 8 : 0);
      hA[((mt * 6 + kt) * 32 + lane2) * 16 + e] = (__bf16)hv;
    }
  }
  __syncthreads();

  // ---- fc1 [64x192]@[192x768] + bias + exact GELU -> gA ----
  for (int idx = wv * 24; idx < wv * 24 + 24; ++idx) {   // 192 tiles
    int mt = idx / 48, nt = idx % 48;
    v8f acc = v8f_zero();
#pragma unroll
    for (int kt = 0; kt < 6; ++kt) {
      v16bf a  = *(const v16bf*)(hA + ((mt * 6 + kt) * 32 + L) * 16);
      v16bf bb = *(const v16bf*)(W1 + ((kt * 48 + nt) * 32 + L) * 16);
      acc = WMMA_BF16(a, bb, acc);
    }
    int c = nt * 16 + (L & 15);
    float bias = b1[c];
    int kt2 = c >> 5, kl = c & 31, rr = kl & 15;
    int e = (rr & 7) + ((kl >= 16) ? 8 : 0);
    int laneAdd = (rr >= 8) ? 16 : 0;
    int hi8 = (L >> 4) << 3;
    __bf16* dst = gA + ((mt * 24 + kt2) * 32 + laneAdd) * 16 + e;
#pragma unroll
    for (int r = 0; r < 8; ++r) {
      float vv = acc[r] + bias;
      float g = 0.5f * vv * (1.f + erff(vv * 0.70710678118654752f));
      dst[(r + hi8) * 16] = (__bf16)g;              // m&15 = r + hi8
    }
  }
  __syncthreads();

  // ---- fc2 [64x768]@[768x192] + bias + residual -> io (in place) ----
  for (int idx = wv * 6; idx < wv * 6 + 6; ++idx) {      // 48 tiles
    int mt = idx / 12, nt = idx % 12;
    v8f acc = v8f_zero();
#pragma unroll 4
    for (int kt = 0; kt < 24; ++kt) {
      v16bf a  = *(const v16bf*)(gA + ((mt * 24 + kt) * 32 + L) * 16);
      v16bf bb = *(const v16bf*)(W2 + ((kt * 12 + nt) * 32 + L) * 16);
      acc = WMMA_BF16(a, bb, acc);
    }
    int c = nt * 16 + (L & 15);
    float bias = b2[c];
    int hi8 = (L >> 4) << 3;
#pragma unroll
    for (int r = 0; r < 8; ++r) {
      int m = mt * 16 + r + hi8;
      io[(base_tok + m) * 192 + c] = acc[r] + bias + xrbuf[m * 192 + c];
    }
  }
}

// ---------------------------------------------------------------------------
extern "C" void kernel_launch(void* const* d_in, const int* in_sizes, int n_in,
                              void* d_out, int out_size, void* d_ws, size_t ws_size,
                              hipStream_t stream) {
  (void)in_sizes; (void)n_in; (void)out_size; (void)ws_size;
  const float* x      = (const float*)d_in[0];
  const float* qkv_w  = (const float*)d_in[1];
  const float* qkv_b  = (const float*)d_in[2];
  const float* proj_w = (const float*)d_in[3];
  const float* proj_b = (const float*)d_in[4];
  const float* n1_g   = (const float*)d_in[5];
  const float* n1_b   = (const float*)d_in[6];
  const float* n2_g   = (const float*)d_in[7];
  const float* n2_b   = (const float*)d_in[8];
  const float* fc1_w  = (const float*)d_in[9];
  const float* fc1_b  = (const float*)d_in[10];
  const float* fc2_w  = (const float*)d_in[11];
  const float* fc2_b  = (const float*)d_in[12];
  const float* amask  = (const float*)d_in[13];
  float* out = (float*)d_out;

  // Workspace: pre-swizzled bf16 weight panels (884736 bytes total).
  __bf16* qkvW  = (__bf16*)d_ws;          // 110592 halves
  __bf16* projW = qkvW + 110592;          //  36864 halves (contiguous after qkvW)
  __bf16* fc1W  = projW + 36864;          // 147456 halves
  __bf16* fc2W  = fc1W + 147456;          // 147456 halves (contiguous after fc1W)

  swizzle_b_kernel<<<432, 256, 0, stream>>>(qkv_w,  qkvW,  576, 36, 110592);
  swizzle_b_kernel<<<144, 256, 0, stream>>>(proj_w, projW, 192, 12,  36864);
  swizzle_b_kernel<<<576, 256, 0, stream>>>(fc1_w,  fc1W,  768, 48, 147456);
  swizzle_b_kernel<<<576, 256, 0, stream>>>(fc2_w,  fc2W,  192, 12, 147456);

  // Attention: one workgroup per window (32 batches * 64 windows).
  swin_attn_kernel<<<2048, 256, 180224, stream>>>(
      x, qkvW, qkv_b, projW, proj_b, n1_g, n1_b, amask, out);

  // MLP: one workgroup per 64-token chunk, in-place on d_out.
  swin_mlp_kernel<<<1568, 256, 172032, stream>>>(
      out, fc1W, fc1_b, fc2W, fc2_b, n2_g, n2_b);
}
